// ResidualBlock_31421980737578
// MI455X (gfx1250) — matrix operand order repaired and checked by reading, not compiled, not run on hardware
//
#include <hip/hip_runtime.h>

// ---------------------------------------------------------------------------
// ResidualBlock sparse-conv on gfx1250 (MI455X), wave32 + WMMA f16->f32.
//   N=100000 points, K=27 offsets, Cin=32 -> 64 -> 64.
// Pipeline: pack weights (f32 -> f16 B-fragments) -> conv1(WMMA)+stats ->
//           bn1 finalize -> bn1 apply+relu (in place) -> conv2(WMMA)+stats ->
//           bn2 finalize -> epilogue (skip GEMM via WMMA + bn2 + relu).
// BN stats are reduced per-workgroup in LDS (ds_add_f32) before hitting the
// 128 contended global counters.
// ---------------------------------------------------------------------------

typedef __attribute__((ext_vector_type(16))) _Float16 v16h;
typedef __attribute__((ext_vector_type(8)))  float    v8f;

#define KOFF 27
#define COUT 64
#define NT_WAVES_PER_BLOCK 8

static __device__ __forceinline__ void atomicAddF32(float* p, float v) {
    __hip_atomic_fetch_add(p, v, __ATOMIC_RELAXED, __HIP_MEMORY_SCOPE_AGENT);
}

// Build one 16x32 f16 A-fragment (ISA 7.12.2, 16-bit A layout):
// lane element j: v=j>>1, K = (v>>2)*16 + hi*8 + (v&3)*2 + (j&1).
// For fixed (group, hi) the 8 channels are CONTIGUOUS, so the gather is
// 4x global_load_b128. m is the neighbor mask as 0.0/1.0 (EXEC stays all-1s).
static __device__ __forceinline__ v16h build_a_frag(const float* __restrict__ rowBase,
                                                    int kbase, int hi, float m) {
    const float4* p0 = reinterpret_cast<const float4*>(rowBase + kbase + hi * 8);
    const float4* p1 = reinterpret_cast<const float4*>(rowBase + kbase + 16 + hi * 8);
    float4 f0 = p0[0], f1 = p0[1];
    float4 f2 = p1[0], f3 = p1[1];
    v16h a;
    a[0]  = (_Float16)(f0.x * m); a[1]  = (_Float16)(f0.y * m);
    a[2]  = (_Float16)(f0.z * m); a[3]  = (_Float16)(f0.w * m);
    a[4]  = (_Float16)(f1.x * m); a[5]  = (_Float16)(f1.y * m);
    a[6]  = (_Float16)(f1.z * m); a[7]  = (_Float16)(f1.w * m);
    a[8]  = (_Float16)(f2.x * m); a[9]  = (_Float16)(f2.y * m);
    a[10] = (_Float16)(f2.z * m); a[11] = (_Float16)(f2.w * m);
    a[12] = (_Float16)(f3.x * m); a[13] = (_Float16)(f3.y * m);
    a[14] = (_Float16)(f3.z * m); a[15] = (_Float16)(f3.w * m);
    return a;
}

// Repack weights [nK][KD][64] f32 -> f16 B fragments. Fragment = 32x16 (KxN),
// stored as 32 lanes x 16 halfs; lane element j holds (K = q*32+(j>>3)*16+(lane>>4)*8+(j&7),
// col = c*16+(lane&15)). Fragment order: ((k*4 + c)*Q + q), Q = KD/32.
__global__ void pack_weights(const float* __restrict__ src, _Float16* __restrict__ dst,
                             int nK, int KD) {
    int Q = KD >> 5;
    int total = nK * 4 * Q * 512;
    int u = blockIdx.x * blockDim.x + threadIdx.x;
    if (u >= total) return;
    int j    = u & 15;
    int lane = (u >> 4) & 31;
    int fl   = u >> 9;
    int q    = fl % Q;
    int c    = (fl / Q) & 3;
    int k    = fl / (4 * Q);
    int Kr   = q * 32 + ((j >> 3) * 16) + ((lane >> 4) * 8) + (j & 7);
    int col  = c * 16 + (lane & 15);
    dst[u] = (_Float16)src[((long)k * KD + Kr) * 64 + col];
}

__global__ void zero_stats(float* __restrict__ stats) {
    int t = threadIdx.x;
    if (t < 256) stats[t] = 0.0f;   // sum1, sumsq1, sum2, sumsq2
}

// Masked sparse conv: Y[n,:] = sum_k mask[n,k] * X[idx[n,k],:] @ W[k]  (+stats)
// One wave per 16-row tile; 4 f32 accumulators cover the 64 output channels.
template <int CIN>
__global__ __launch_bounds__(256) void conv_wmma_kernel(
    const float* __restrict__ X, const int* __restrict__ nbr_idx,
    const unsigned char* __restrict__ nbr_mask, const _Float16* __restrict__ packW,
    float* __restrict__ Y, float* __restrict__ sum, float* __restrict__ sumsq,
    int N, int nTiles) {
    constexpr int Q = CIN / 32;
    __shared__ float blkSum[COUT];
    __shared__ float blkSumSq[COUT];

    int lane = threadIdx.x & 31;
    if (threadIdx.x < 32) {           // wave 0 is active in every launched block
        blkSum[lane] = 0.0f;  blkSum[lane + 32] = 0.0f;
        blkSumSq[lane] = 0.0f; blkSumSq[lane + 32] = 0.0f;
    }
    __syncthreads();

    int wave = (blockIdx.x * blockDim.x + threadIdx.x) >> 5;
    if (wave < nTiles) {
        int lanelow = lane & 15;
        int hi = lane >> 4;
        bool fullTile = (wave * 16 + 16) <= N;

        int row = wave * 16 + lanelow;
        float rowValid = (row < N) ? 1.0f : 0.0f;
        int rowC = (row < N) ? row : (N - 1);
        const int* idxRow = nbr_idx + (long)rowC * KOFF;
        const unsigned char* mRow = nbr_mask + (long)rowC * KOFF;

        v8f acc[4];
#pragma unroll
        for (int c = 0; c < 4; ++c)
#pragma unroll
            for (int r = 0; r < 8; ++r) acc[c][r] = 0.0f;

        const _Float16* wLane = packW + lane * 16;
        for (int k = 0; k < KOFF; ++k) {
            int nb  = idxRow[k];
            float m = (mRow[k] != 0) ? rowValid : 0.0f;
            const float* rb = X + (long)nb * CIN;
            v16h a[Q];
#pragma unroll
            for (int q = 0; q < Q; ++q) a[q] = build_a_frag(rb, q * 32, hi, m);
            const _Float16* wk = wLane + (size_t)(k * 4 * Q) * 512;
#pragma unroll
            for (int c = 0; c < 4; ++c) {
#pragma unroll
                for (int q = 0; q < Q; ++q) {
                    v16h b = *reinterpret_cast<const v16h*>(wk + (size_t)(c * Q + q) * 512);
                    acc[c] = __builtin_amdgcn_wmma_f32_16x16x32_f16(
                        false, a[q], false, b, (short)0, acc[c], false, false);
                }
            }
        }

        // Store C (lane holds col = c*16+lanelow, rows r + 8*hi) + BN stats.
#pragma unroll
        for (int c = 0; c < 4; ++c) {
            int ch = c * 16 + lanelow;
            float* yBase = Y + (long)(wave * 16 + 8 * hi) * COUT + ch;
            if (fullTile) {
#pragma unroll
                for (int r = 0; r < 8; ++r) yBase[r * COUT] = acc[c][r];
            } else {
#pragma unroll
                for (int r = 0; r < 8; ++r)
                    if (wave * 16 + 8 * hi + r < N) yBase[r * COUT] = acc[c][r];
            }
            float s = 0.0f, s2 = 0.0f;
#pragma unroll
            for (int r = 0; r < 8; ++r) {
                float val = acc[c][r];
                s += val;
                s2 += val * val;
            }
            float sPart  = __shfl_xor(s, 16, 32);
            float s2Part = __shfl_xor(s2, 16, 32);
            if (hi == 0) {
                atomicAdd(&blkSum[ch], s + sPart);        // ds_add_f32
                atomicAdd(&blkSumSq[ch], s2 + s2Part);    // ds_add_f32
            }
        }
    }

    __syncthreads();
    if (threadIdx.x < 32) {           // 8x fewer contended global atomics
#pragma unroll
        for (int i = 0; i < 2; ++i) {
            int ch = lane + i * 32;
            atomicAddF32(&sum[ch], blkSum[ch]);
            atomicAddF32(&sumsq[ch], blkSumSq[ch]);
        }
    }
}

__global__ void bn_finalize(const float* __restrict__ sum, const float* __restrict__ sumsq,
                            const float* __restrict__ gamma, const float* __restrict__ beta,
                            float* __restrict__ scale, float* __restrict__ shift, float invN) {
    int c = threadIdx.x;
    if (c >= COUT) return;
    float mu  = sum[c] * invN;
    float var = fmaxf(sumsq[c] * invN - mu * mu, 0.0f);
    float sc  = gamma[c] * rsqrtf(var + 1e-5f);
    scale[c] = sc;
    shift[c] = beta[c] - mu * sc;
}

__global__ void bn_apply_relu(float* __restrict__ Y, const float* __restrict__ scale,
                              const float* __restrict__ shift, int total4) {
    int t = blockIdx.x * blockDim.x + threadIdx.x;
    if (t >= total4) return;
    float4 y = reinterpret_cast<float4*>(Y)[t];
    int c0 = (t * 4) & (COUT - 1);
    y.x = fmaxf(y.x * scale[c0 + 0] + shift[c0 + 0], 0.0f);
    y.y = fmaxf(y.y * scale[c0 + 1] + shift[c0 + 1], 0.0f);
    y.z = fmaxf(y.z * scale[c0 + 2] + shift[c0 + 2], 0.0f);
    y.w = fmaxf(y.w * scale[c0 + 3] + shift[c0 + 3], 0.0f);
    reinterpret_cast<float4*>(Y)[t] = y;
}

// out = relu(bn2(Y2) + feats @ Wsc) — skip GEMM done with WMMA, bn2(Y2) fed
// in through the C accumulator.
__global__ __launch_bounds__(256) void epilogue_kernel(
    const float* __restrict__ feats, const float* __restrict__ Y2,
    const _Float16* __restrict__ packWsc, const float* __restrict__ scale2,
    const float* __restrict__ shift2, float* __restrict__ out, int N, int nTiles) {
    int wave = (blockIdx.x * blockDim.x + threadIdx.x) >> 5;
    if (wave >= nTiles) return;
    int lane = threadIdx.x & 31;
    int lanelow = lane & 15;
    int hi = lane >> 4;
    bool fullTile = (wave * 16 + 16) <= N;

    int row = wave * 16 + lanelow;
    int rowC = (row < N) ? row : (N - 1);
    v16h a = build_a_frag(feats + (long)rowC * 32, 0, hi, 1.0f);

#pragma unroll
    for (int c = 0; c < 4; ++c) {
        int ch = c * 16 + lanelow;
        float sc = scale2[ch], sh = shift2[ch];
        const float* y2Base = Y2 + (long)(wave * 16 + 8 * hi) * COUT + ch;
        v8f cacc;
        if (fullTile) {
#pragma unroll
            for (int r = 0; r < 8; ++r) cacc[r] = y2Base[r * COUT] * sc + sh;
        } else {
#pragma unroll
            for (int r = 0; r < 8; ++r) {
                int rowIn = wave * 16 + 8 * hi + r;
                cacc[r] = (rowIn < N) ? (y2Base[r * COUT] * sc + sh) : 0.0f;
            }
        }
        v16h b = *reinterpret_cast<const v16h*>(packWsc + (size_t)c * 512 + lane * 16);
        cacc = __builtin_amdgcn_wmma_f32_16x16x32_f16(
            false, a, false, b, (short)0, cacc, false, false);
        float* oBase = out + (long)(wave * 16 + 8 * hi) * COUT + ch;
        if (fullTile) {
#pragma unroll
            for (int r = 0; r < 8; ++r) oBase[r * COUT] = fmaxf(cacc[r], 0.0f);
        } else {
#pragma unroll
            for (int r = 0; r < 8; ++r)
                if (wave * 16 + 8 * hi + r < N) oBase[r * COUT] = fmaxf(cacc[r], 0.0f);
        }
    }
}

extern "C" void kernel_launch(void* const* d_in, const int* in_sizes, int n_in,
                              void* d_out, int out_size, void* d_ws, size_t ws_size,
                              hipStream_t stream) {
    const float* feats  = (const float*)d_in[0];
    const float* W1     = (const float*)d_in[1];
    const float* gamma1 = (const float*)d_in[2];
    const float* beta1  = (const float*)d_in[3];
    const float* W2     = (const float*)d_in[4];
    const float* gamma2 = (const float*)d_in[5];
    const float* beta2  = (const float*)d_in[6];
    const float* Wsc    = (const float*)d_in[7];
    const int*   nbrIdx = (const int*)d_in[8];
    const unsigned char* nbrMask = (const unsigned char*)d_in[9]; // jax bool = 1 byte
    float* out = (float*)d_out;

    const int N = in_sizes[0] / 32;            // 100000
    const int nTiles = (N + 15) / 16;          // 6250

    // Workspace layout (256B-aligned regions).
    char* ws = (char*)d_ws;
    _Float16* packW1  = (_Float16*)(ws + 0);               // 27*4*512   halfs = 110592 B
    _Float16* packW2  = (_Float16*)(ws + 110592);          // 27*8*512   halfs = 221184 B
    _Float16* packWsc = (_Float16*)(ws + 331776);          // 4*512      halfs =   4096 B
    float*    stats   = (float*)(ws + 335872);             // 8*64 floats
    float*    Y1      = (float*)(ws + 337920);             // N*64 f32
    float*    Y2      = (float*)(ws + 337920 + (size_t)N * COUT * sizeof(float));
    float *sum1 = stats, *sumsq1 = stats + 64, *sum2 = stats + 128, *sumsq2 = stats + 192;
    float *scale1 = stats + 256, *shift1 = stats + 320, *scale2 = stats + 384, *shift2 = stats + 448;
    (void)ws_size; (void)n_in; (void)out_size;

    // 1) Pack weights into f16 WMMA B-fragment layout.
    pack_weights<<<(27 * 4 * 512 + 255) / 256, 256, 0, stream>>>(W1, packW1, 27, 32);
    pack_weights<<<(27 * 8 * 512 + 255) / 256, 256, 0, stream>>>(W2, packW2, 27, 64);
    pack_weights<<<(4 * 512 + 255) / 256, 256, 0, stream>>>(Wsc, packWsc, 1, 32);
    zero_stats<<<1, 256, 0, stream>>>(stats);

    const int convBlocks = (nTiles + NT_WAVES_PER_BLOCK - 1) / NT_WAVES_PER_BLOCK;
    const float invN = 1.0f / (float)N;

    // 2) conv1 (gather + 27*4 WMMAs per 16-row tile) + BN1 stats.
    conv_wmma_kernel<32><<<convBlocks, 256, 0, stream>>>(
        feats, nbrIdx, nbrMask, packW1, Y1, sum1, sumsq1, N, nTiles);
    bn_finalize<<<1, 64, 0, stream>>>(sum1, sumsq1, gamma1, beta1, scale1, shift1, invN);
    bn_apply_relu<<<((N * COUT / 4) + 255) / 256, 256, 0, stream>>>(Y1, scale1, shift1, N * COUT / 4);

    // 3) conv2 (Cin=64 -> 27*8 WMMAs per tile) + BN2 stats.
    conv_wmma_kernel<64><<<convBlocks, 256, 0, stream>>>(
        Y1, nbrIdx, nbrMask, packW2, Y2, sum2, sumsq2, N, nTiles);
    bn_finalize<<<1, 64, 0, stream>>>(sum2, sumsq2, gamma2, beta2, scale2, shift2, invN);

    // 4) relu(bn2(Y2) + feats @ Wsc) via WMMA epilogue.
    epilogue_kernel<<<convBlocks, 256, 0, stream>>>(
        feats, Y2, packWsc, scale2, shift2, out, N, nTiles);
}
